// PointNetSetAbstraction_39779987095778
// MI455X (gfx1250) — matrix operand order repaired and checked by reading, not compiled, not run on hardware
//
#include <hip/hip_runtime.h>

// ---------------------------------------------------------------------------
// PointNet++ SetAbstraction for MI455X (gfx1250, wave32, WMMA 16x16x32 f16)
// f16 activations live in a ZERO-PADDED layout [B, 34, 514, C] so the 3x3
// implicit-GEMM WMMA kernel needs no boundary predication at all.
// f32 conv outputs stay compact [B*P, C].  BN = training-mode stats.
// GEMM: one wave computes 16(Cout) x 64(pos), 4 accumulators sharing A frag.
// ---------------------------------------------------------------------------

typedef _Float16 h8   __attribute__((ext_vector_type(8)));
typedef _Float16 v16h __attribute__((ext_vector_type(16)));
typedef float    v8f  __attribute__((ext_vector_type(8)));
typedef float    f4   __attribute__((ext_vector_type(4)));

#define BATCH   16
#define NPTS    4096
#define NPT     512      // npoint (W)
#define NSAMP   32       // nsample (H)
#define PPOS    (NSAMP * NPT)     // 16384 compact positions per batch image
#define BP      (BATCH * PPOS)    // 262144 total compact positions
#define HP      34                // padded H
#define WP      514               // padded W
#define PPAD    (HP * WP)         // 17476 padded positions per batch image
#define R2      0.04f

// -------------------------- FPS (one block / batch) -------------------------
__global__ __launch_bounds__(256) void fps_kernel(const float* __restrict__ xyz,
                                                  float* __restrict__ nxyz) {
  __shared__ float dist[NPTS];
  __shared__ float rv[256];
  __shared__ int   ri[256];
  __shared__ float cent[3];
  __shared__ int   sfar;
  const int t = threadIdx.x, b = blockIdx.x;
  const float* xb = xyz + (size_t)b * 3 * NPTS;
  for (int n = t; n < NPTS; n += 256) dist[n] = 1e10f;
  if (t == 0) sfar = 0;
  __syncthreads();
  for (int it = 0; it < NPT; ++it) {
    int far = sfar;
    if (t < 3) {
      float v = xb[t * NPTS + far];
      cent[t] = v;
      nxyz[((size_t)b * NPT + it) * 3 + t] = v;  // emit OLD far (matches scan)
    }
    __syncthreads();
    float cx = cent[0], cy = cent[1], cz = cent[2];
    float bm = -1.f; int bi = 0;
    for (int n = t; n < NPTS; n += 256) {
      float dx = xb[n] - cx, dy = xb[NPTS + n] - cy, dz = xb[2 * NPTS + n] - cz;
      float d = dx * dx + dy * dy + dz * dz;
      float nd = dist[n]; nd = nd < d ? nd : d; dist[n] = nd;
      if (nd > bm) { bm = nd; bi = n; }   // ascending n => first-occurrence tie
    }
    rv[t] = bm; ri[t] = bi;
    __syncthreads();
    for (int s2 = 128; s2 > 0; s2 >>= 1) {
      if (t < s2) {
        float ov = rv[t + s2]; int oi = ri[t + s2];
        if (ov > rv[t] || (ov == rv[t] && oi < ri[t])) { rv[t] = ov; ri[t] = oi; }
      }
      __syncthreads();
    }
    if (t == 0) sfar = ri[0];
    __syncthreads();
  }
}

// new_xyz [B,S,3] -> d_out [B,3,S]
__global__ void nxyz_out_kernel(const float* __restrict__ nxyz, float* __restrict__ out) {
  int t = blockIdx.x * blockDim.x + threadIdx.x;          // 16*3*512
  int s = t & 511; int c = (t >> 9) % 3; int b = t / (512 * 3);
  out[((size_t)b * 3 + c) * NPT + s] = nxyz[((size_t)b * NPT + s) * 3 + c];
}

// ------------------------------ Ball query ----------------------------------
__global__ __launch_bounds__(256) void ball_query_kernel(const float* __restrict__ xyz,
                                                         const float* __restrict__ nxyz,
                                                         int* __restrict__ grp) {
  int t = blockIdx.x * blockDim.x + threadIdx.x;          // B*NPT = 8192
  int s = t & 511; int b = t >> 9;
  const float* xb = xyz + (size_t)b * 3 * NPTS;
  const float* nz = nxyz + ((size_t)b * NPT + s) * 3;
  float cx = nz[0], cy = nz[1], cz = nz[2];
  int* g = grp + ((size_t)b * NPT + s) * NSAMP;
  int cnt = 0, first = 0;
  for (int n = 0; n < NPTS && cnt < NSAMP; ++n) {
    float dx = xb[n] - cx, dy = xb[NPTS + n] - cy, dz = xb[2 * NPTS + n] - cz;
    float d = dx * dx + dy * dy + dz * dz;
    if (d <= R2) { if (cnt == 0) first = n; g[cnt++] = n; }
  }
  for (; cnt < NSAMP; ++cnt) g[cnt] = first;
}

// -------- grouping -> x0 f16 [B*P, 6]; position p = h*512 + w (h=sample) ----
__global__ __launch_bounds__(256) void group_kernel(const float* __restrict__ xyz,
                                                    const float* __restrict__ pts,
                                                    const float* __restrict__ nxyz,
                                                    const int* __restrict__ grp,
                                                    _Float16* __restrict__ x0) {
  int t = blockIdx.x * blockDim.x + threadIdx.x;          // BP
  int j = t & 31; int s = (t >> 5) & 511; int b = t >> 14;
  int id = grp[(((size_t)b * NPT + s) * NSAMP) + j];
  int p = j * NPT + s;
  size_t row = ((size_t)b * PPOS + p) * 6;
  const float* xb = xyz + (size_t)b * 3 * NPTS;
  const float* pb = pts + (size_t)b * 3 * NPTS;
  const float* nz = nxyz + ((size_t)b * NPT + s) * 3;
  x0[row + 0] = (_Float16)(xb[id] - nz[0]);
  x0[row + 1] = (_Float16)(xb[NPTS + id] - nz[1]);
  x0[row + 2] = (_Float16)(xb[2 * NPTS + id] - nz[2]);
  x0[row + 3] = (_Float16)pb[id];
  x0[row + 4] = (_Float16)pb[NPTS + id];
  x0[row + 5] = (_Float16)pb[2 * NPTS + id];
}

// ----------------------- conv1 (6->64, K too small for WMMA) ----------------
// writes compact f32 (for residual) and padded f16 (for WMMA conv_a)
__global__ __launch_bounds__(256) void conv1_kernel(const _Float16* __restrict__ x0,
                                                    const float* __restrict__ w,
                                                    const float* __restrict__ bias,
                                                    float* __restrict__ h1f,
                                                    _Float16* __restrict__ h1p) {
  int t = blockIdx.x * blockDim.x + threadIdx.x;          // BP
  size_t r6 = (size_t)t * 6;
  float xin[6];
#pragma unroll
  for (int i = 0; i < 6; ++i) xin[i] = (float)x0[r6 + i];
  int b = t >> 14; int p = t & (PPOS - 1); int h = p >> 9; int wq = p & 511;
  size_t r64 = (size_t)t * 64;
  size_t rp  = ((size_t)b * PPAD + (size_t)(h + 1) * WP + (wq + 1)) * 64;
  for (int co = 0; co < 64; ++co) {
    float a = bias[co];
#pragma unroll
    for (int i = 0; i < 6; ++i) a += w[co * 6 + i] * xin[i];
    h1f[r64 + co] = a;
    h1p[rp + co]  = (_Float16)a;
  }
}

// ------------------- WMMA implicit-GEMM conv (1x1 or 3x3 SAME) --------------
// act  : f16 PADDED [B, HP*WP, CIN]   wgt : f16 [COUT, TAPS*CIN]
// out  : f32 compact [B*P, COUT]      TAPS: 1 or 9 (dy=tap/3-1, dx=tap%3-1)
// One wave computes a 16(Cout) x 64(pos) tile: 4 N-subtiles share the A frag.
// All loads are unconditional: the halo of `act` is pre-zeroed.
template<int CIN, int COUT, int TAPS>
__global__ __launch_bounds__(256) void wmma_conv_kernel(const _Float16* __restrict__ act,
                                                        const _Float16* __restrict__ wgt,
                                                        const float* __restrict__ bias,
                                                        float* __restrict__ out) {
  const int lane  = threadIdx.x & 31;
  const int wave  = blockIdx.x * (blockDim.x >> 5) + (threadIdx.x >> 5);
  const int tilesN = BP / 64;                 // 4096 position macro-tiles
  const int tileN = wave & (tilesN - 1);
  const int tileM = wave / tilesN;
  const int gp0 = tileN * 64;
  const int b   = gp0 >> 14;                  // PPOS = 16384
  const int pp  = gp0 & (PPOS - 1);
  const int h0  = pp >> 9;
  const int w0  = pp & 511;                   // multiple of 64
  const int co0 = tileM * 16;
  const int ln15 = lane & 15;
  const int hi   = lane >> 4;
  const int kb   = hi * 8;                    // A-frag K sub-offset
  const int klo  = hi * 8;                    // B-frag K sub-offset
  constexpr int Ktot = TAPS * CIN;
  const _Float16* wrow  = wgt + (size_t)(co0 + ln15) * Ktot;
  const _Float16* abase = act + (size_t)b * PPAD * CIN;
  v8f acc[4] = {{}, {}, {}, {}};
#pragma unroll
  for (int tap = 0; tap < TAPS; ++tap) {
    const int dy = (TAPS == 9) ? (tap / 3 - 1) : 0;
    const int dx = (TAPS == 9) ? (tap % 3 - 1) : 0;
    const int hh = h0 + dy + 1;               // padded row, always in [0,HP)
    const _Float16* arow = wrow + tap * CIN;
#pragma unroll
    for (int kc = 0; kc < CIN; kc += 32) {
      v16h afrag;
      {
        h8 alo = *(const h8*)(arow + kc + kb);
        h8 ahi = *(const h8*)(arow + kc + 16 + kb);
#pragma unroll
        for (int i = 0; i < 8; ++i) { afrag[i] = alo[i]; afrag[8 + i] = ahi[i]; }
      }
#pragma unroll
      for (int sub = 0; sub < 4; ++sub) {
        const int ww = w0 + sub * 16 + ln15 + dx + 1;   // padded col, in [0,WP)
        const _Float16* brow = abase + ((size_t)hh * WP + ww) * CIN;
        h8 blo = *(const h8*)(brow + kc + klo);
        h8 bhi = *(const h8*)(brow + kc + 16 + klo);
        v16h bfrag;
#pragma unroll
        for (int i = 0; i < 8; ++i) { bfrag[i] = blo[i]; bfrag[8 + i] = bhi[i]; }
        acc[sub] = __builtin_amdgcn_wmma_f32_16x16x32_f16(
            false, afrag, false, bfrag, (short)0, acc[sub], false, false);
      }
    }
  }
#pragma unroll
  for (int sub = 0; sub < 4; ++sub) {
#pragma unroll
    for (int r = 0; r < 8; ++r) acc[sub][r] += bias[co0 + hi * 8 + r];
    float* orow = out + (((size_t)b * PPOS + pp + sub * 16 + ln15) * COUT) + co0 + hi * 8;
    f4 o0 = {acc[sub][0], acc[sub][1], acc[sub][2], acc[sub][3]};
    f4 o1 = {acc[sub][4], acc[sub][5], acc[sub][6], acc[sub][7]};
    *(f4*)(orow)     = o0;
    *(f4*)(orow + 4) = o1;
  }
}

// --------------------------- BatchNorm helpers ------------------------------
// C is a power of two and divides 256, so each thread owns one fixed channel.
__global__ __launch_bounds__(256) void bn_stats_kernel(const float* __restrict__ x,
                                                       float* __restrict__ stats,
                                                       int C, size_t total) {
  __shared__ float ls[256];
  int t = threadIdx.x;
  if (t < 2 * C) ls[t] = 0.f;
  __syncthreads();
  float s = 0.f, q = 0.f;
  size_t stride = (size_t)gridDim.x * blockDim.x;
  for (size_t i = (size_t)blockIdx.x * blockDim.x + t; i < total; i += stride) {
    float v = x[i]; s += v; q += v * v;
  }
  int c = t & (C - 1);
  atomicAdd(&ls[c], s);
  atomicAdd(&ls[C + c], q);
  __syncthreads();
  if (t < 2 * C) atomicAdd(&stats[t], ls[t]);
}

__global__ void bn_finalize_kernel(const float* __restrict__ stats,
                                   const float* __restrict__ g,
                                   const float* __restrict__ beta,
                                   float* __restrict__ ss, int C, float invN) {
  int c = threadIdx.x;
  if (c < C) {
    float mean = stats[c] * invN;
    float var  = stats[C + c] * invN - mean * mean;
    float sc   = g[c] * rsqrtf(var + 1e-5f);
    ss[c]      = sc;
    ss[C + c]  = beta[c] - mean * sc;
  }
}

// y (PADDED f16) = relu(bn(x compact f32))
__global__ __launch_bounds__(256) void bn_apply_relu_pad_kernel(const float* __restrict__ x,
                                                                _Float16* __restrict__ y,
                                                                const float* __restrict__ ss,
                                                                int C, int cshift, size_t total) {
  size_t i = (size_t)blockIdx.x * blockDim.x + threadIdx.x;
  if (i < total) {
    int c = (int)(i & (size_t)(C - 1));
    size_t pos = i >> cshift;
    int b = (int)(pos >> 14);
    int p = (int)(pos & (PPOS - 1));
    int h = p >> 9, wq = p & 511;
    float v = x[i] * ss[c] + ss[C + c];
    y[((size_t)b * PPAD + (size_t)(h + 1) * WP + (wq + 1)) * C + c] =
        (_Float16)(v > 0.f ? v : 0.f);
  }
}

// x = bn(x) + res   (in place, compact f32)
__global__ __launch_bounds__(256) void bn_residual_kernel(float* __restrict__ x,
                                                          const float* __restrict__ res,
                                                          const float* __restrict__ ss,
                                                          int C, size_t total) {
  size_t i = (size_t)blockIdx.x * blockDim.x + threadIdx.x;
  if (i < total) {
    int c = (int)(i & (size_t)(C - 1));
    x[i] = x[i] * ss[c] + ss[C + c] + res[i];
  }
}

// out[b,co,w] = max_h relu(bn3(c3[b, h*512+w, co]))
__global__ __launch_bounds__(256) void bn_max_kernel(const float* __restrict__ c3,
                                                     const float* __restrict__ ss,
                                                     float* __restrict__ out) {
  int t = blockIdx.x * blockDim.x + threadIdx.x;          // 16*128*512
  int co = t & 127;
  int w  = (t >> 7) & 511;
  int b  = t >> 16;
  float sc = ss[co], sh = ss[128 + co];
  float m = 0.f;
  const float* base = c3 + ((size_t)b * PPOS + w) * 128 + co;
  for (int h = 0; h < NSAMP; ++h) {
    float v = base[(size_t)h * NPT * 128];
    float y = v * sc + sh; y = y > 0.f ? y : 0.f;
    m = m > y ? m : y;
  }
  out[((size_t)b * 128 + co) * NPT + w] = m;
}

// ----------------------------- weight prep ----------------------------------
__global__ void prep_w1x1_kernel(const float* __restrict__ w, _Float16* __restrict__ o, int n) {
  int i = blockIdx.x * blockDim.x + threadIdx.x;
  if (i < n) o[i] = (_Float16)w[i];
}
// [64,64,3,3] -> f16 [co][tap*64 + ci]
__global__ void prep_w3x3_kernel(const float* __restrict__ w, _Float16* __restrict__ o) {
  int i = blockIdx.x * blockDim.x + threadIdx.x;          // 64*64*9
  if (i < 64 * 64 * 9) {
    int tap = i % 9; int ci = (i / 9) & 63; int co = i / (9 * 64);
    o[(size_t)co * 576 + tap * 64 + ci] = (_Float16)w[i];
  }
}

// ---------------------------------------------------------------------------
extern "C" void kernel_launch(void* const* d_in, const int* in_sizes, int n_in,
                              void* d_out, int out_size, void* d_ws, size_t ws_size,
                              hipStream_t stream) {
  (void)in_sizes; (void)n_in; (void)out_size; (void)ws_size;
  // pytree order: xyz, points, then params flattened with dict keys sorted:
  // bns[i]{beta,g}, bott[j]{b,beta,g,w}, convs[i]{b,w}
  const float* xyz      = (const float*)d_in[0];
  const float* pts      = (const float*)d_in[1];
  const float* bns0_beta = (const float*)d_in[2];
  const float* bns0_g    = (const float*)d_in[3];
  const float* bns1_beta = (const float*)d_in[4];
  const float* bns1_g    = (const float*)d_in[5];
  const float* bns2_beta = (const float*)d_in[6];
  const float* bns2_g    = (const float*)d_in[7];
  const float* ba_b = (const float*)d_in[8];   const float* ba_beta = (const float*)d_in[9];
  const float* ba_g = (const float*)d_in[10];  const float* ba_w    = (const float*)d_in[11];
  const float* bb_b = (const float*)d_in[12];  const float* bb_beta = (const float*)d_in[13];
  const float* bb_g = (const float*)d_in[14];  const float* bb_w    = (const float*)d_in[15];
  const float* bc_b = (const float*)d_in[16];  const float* bc_beta = (const float*)d_in[17];
  const float* bc_g = (const float*)d_in[18];  const float* bc_w    = (const float*)d_in[19];
  const float* bd_b = (const float*)d_in[20];  const float* bd_beta = (const float*)d_in[21];
  const float* bd_g = (const float*)d_in[22];  const float* bd_w    = (const float*)d_in[23];
  const float* c0_b = (const float*)d_in[24];  const float* c0_w    = (const float*)d_in[25];
  const float* c1_b = (const float*)d_in[26];  const float* c1_w    = (const float*)d_in[27];
  const float* c2_b = (const float*)d_in[28];  const float* c2_w    = (const float*)d_in[29];
  float* out = (float*)d_out;

  // ---- workspace layout (c3 aliases h1f32 region, dead after residual) ----
  char* Wk = (char*)d_ws;
  size_t off = 0;
  auto alloc = [&](size_t bytes) { void* p = Wk + off; off += (bytes + 255) & ~(size_t)255; return p; };
  const size_t padB64 = (size_t)BATCH * PPAD * 64 * 2;
  const size_t padB32 = (size_t)BATCH * PPAD * 32 * 2;
  float*    bufA  = (float*)alloc((size_t)BP * 128 * 4);   // h1_f32 [BP,64] then c3 [BP,128]
  float*    h1f32 = bufA;
  float*    c3    = bufA;
  float*    cbuf  = (float*)alloc((size_t)BP * 64 * 4);    // c_a / c_c / c_d+h / c2
  _Float16* actAp = (_Float16*)alloc(padB64);              // padded t1 / t3 / x1 / x2
  _Float16* h1p   = (_Float16*)alloc(padB64);              // padded conv1 output
  float*    cb32  = (float*)alloc((size_t)BP * 32 * 4);    // c_b
  _Float16* actBp = (_Float16*)alloc(padB32);              // padded t2
  _Float16* x0    = (_Float16*)alloc((size_t)BP * 6 * 2);
  float*    nxyz  = (float*)alloc((size_t)BATCH * NPT * 3 * 4);
  int*      grp   = (int*)alloc((size_t)BATCH * NPT * NSAMP * 4);
  _Float16* wAa   = (_Float16*)alloc(64 * 576 * 2);
  _Float16* wAd   = (_Float16*)alloc(64 * 576 * 2);
  _Float16* wAb   = (_Float16*)alloc(32 * 64 * 2);
  _Float16* wAc   = (_Float16*)alloc(64 * 32 * 2);
  _Float16* wA2   = (_Float16*)alloc(64 * 64 * 2);
  _Float16* wA3   = (_Float16*)alloc(128 * 64 * 2);
  float*    stats = (float*)alloc(7 * 256 * 4);
  float*    ss    = (float*)alloc(7 * 256 * 4);

  const float invN = 1.f / (float)BP;
  // zero BN accumulators and activation halos (ws is poisoned, not re-zeroed)
  hipMemsetAsync(stats, 0, 7 * 256 * 4, stream);
  hipMemsetAsync(actAp, 0, padB64, stream);
  hipMemsetAsync(h1p,   0, padB64, stream);
  hipMemsetAsync(actBp, 0, padB32, stream);

  // weight prep
  prep_w3x3_kernel<<<144, 256, 0, stream>>>(ba_w, wAa);
  prep_w3x3_kernel<<<144, 256, 0, stream>>>(bd_w, wAd);
  prep_w1x1_kernel<<<8, 256, 0, stream>>>(bb_w, wAb, 32 * 64);
  prep_w1x1_kernel<<<8, 256, 0, stream>>>(bc_w, wAc, 64 * 32);
  prep_w1x1_kernel<<<16, 256, 0, stream>>>(c1_w, wA2, 64 * 64);
  prep_w1x1_kernel<<<32, 256, 0, stream>>>(c2_w, wA3, 128 * 64);

  // sampling / grouping
  fps_kernel<<<BATCH, 256, 0, stream>>>(xyz, nxyz);
  nxyz_out_kernel<<<(BATCH * 3 * NPT) / 256, 256, 0, stream>>>(nxyz, out);
  ball_query_kernel<<<(BATCH * NPT) / 256, 256, 0, stream>>>(xyz, nxyz, grp);
  group_kernel<<<BP / 256, 256, 0, stream>>>(xyz, pts, nxyz, grp, x0);

  // conv1: 6 -> 64
  conv1_kernel<<<BP / 256, 256, 0, stream>>>(x0, c0_w, c0_b, h1f32, h1p);

  const size_t tot64 = (size_t)BP * 64, tot32 = (size_t)BP * 32, tot128 = (size_t)BP * 128;
  // waves = (COUT/16) * (BP/64); blocks = waves / 8
  const int blk_co64 = (4 * (BP / 64)) / 8;   // 2048
  const int blk_co32 = (2 * (BP / 64)) / 8;   // 1024
  const int blk_co128 = (8 * (BP / 64)) / 8;  // 4096

  // bottleneck conv_a: 3x3 64->64 (WMMA), BN a, relu
  wmma_conv_kernel<64, 64, 9><<<blk_co64, 256, 0, stream>>>(h1p, wAa, ba_b, cbuf);
  bn_stats_kernel<<<1024, 256, 0, stream>>>(cbuf, stats + 0 * 256, 64, tot64);
  bn_finalize_kernel<<<1, 128, 0, stream>>>(stats + 0 * 256, ba_g, ba_beta, ss + 0 * 256, 64, invN);
  bn_apply_relu_pad_kernel<<<tot64 / 256, 256, 0, stream>>>(cbuf, actAp, ss + 0 * 256, 64, 6, tot64);

  // conv_b: 1x1 64->32
  wmma_conv_kernel<64, 32, 1><<<blk_co32, 256, 0, stream>>>(actAp, wAb, bb_b, cb32);
  bn_stats_kernel<<<1024, 256, 0, stream>>>(cb32, stats + 1 * 256, 32, tot32);
  bn_finalize_kernel<<<1, 128, 0, stream>>>(stats + 1 * 256, bb_g, bb_beta, ss + 1 * 256, 32, invN);
  bn_apply_relu_pad_kernel<<<tot32 / 256, 256, 0, stream>>>(cb32, actBp, ss + 1 * 256, 32, 5, tot32);

  // conv_c: 1x1 32->64
  wmma_conv_kernel<32, 64, 1><<<blk_co64, 256, 0, stream>>>(actBp, wAc, bc_b, cbuf);
  bn_stats_kernel<<<1024, 256, 0, stream>>>(cbuf, stats + 2 * 256, 64, tot64);
  bn_finalize_kernel<<<1, 128, 0, stream>>>(stats + 2 * 256, bc_g, bc_beta, ss + 2 * 256, 64, invN);
  bn_apply_relu_pad_kernel<<<tot64 / 256, 256, 0, stream>>>(cbuf, actAp, ss + 2 * 256, 64, 6, tot64);

  // conv_d: 3x3 64->64, BN d (no relu), + residual h1
  wmma_conv_kernel<64, 64, 9><<<blk_co64, 256, 0, stream>>>(actAp, wAd, bd_b, cbuf);
  bn_stats_kernel<<<1024, 256, 0, stream>>>(cbuf, stats + 3 * 256, 64, tot64);
  bn_finalize_kernel<<<1, 128, 0, stream>>>(stats + 3 * 256, bd_g, bd_beta, ss + 3 * 256, 64, invN);
  bn_residual_kernel<<<tot64 / 256, 256, 0, stream>>>(cbuf, h1f32, ss + 3 * 256, 64, tot64);

  // bns[0] + relu -> x1
  bn_stats_kernel<<<1024, 256, 0, stream>>>(cbuf, stats + 4 * 256, 64, tot64);
  bn_finalize_kernel<<<1, 128, 0, stream>>>(stats + 4 * 256, bns0_g, bns0_beta, ss + 4 * 256, 64, invN);
  bn_apply_relu_pad_kernel<<<tot64 / 256, 256, 0, stream>>>(cbuf, actAp, ss + 4 * 256, 64, 6, tot64);

  // conv2: 1x1 64->64, BN, relu -> x2
  wmma_conv_kernel<64, 64, 1><<<blk_co64, 256, 0, stream>>>(actAp, wA2, c1_b, cbuf);
  bn_stats_kernel<<<1024, 256, 0, stream>>>(cbuf, stats + 5 * 256, 64, tot64);
  bn_finalize_kernel<<<1, 128, 0, stream>>>(stats + 5 * 256, bns1_g, bns1_beta, ss + 5 * 256, 64, invN);
  bn_apply_relu_pad_kernel<<<tot64 / 256, 256, 0, stream>>>(cbuf, actAp, ss + 5 * 256, 64, 6, tot64);

  // conv3: 1x1 64->128 (c3 aliases h1f32 region, now dead)
  wmma_conv_kernel<64, 128, 1><<<blk_co128, 256, 0, stream>>>(actAp, wA3, c2_b, c3);
  bn_stats_kernel<<<1024, 256, 0, stream>>>(c3, stats + 6 * 256, 128, tot128);
  bn_finalize_kernel<<<1, 128, 0, stream>>>(stats + 6 * 256, bns2_g, bns2_beta, ss + 6 * 256, 128, invN);
  bn_max_kernel<<<(BATCH * 128 * NPT) / 256, 256, 0, stream>>>(c3, ss + 6 * 256, out + BATCH * 3 * NPT);
}
